// AnalogLSTM_36189394436126
// MI455X (gfx1250) — matrix operand order repaired and checked by reading, not compiled, not run on hardware
//
#include <hip/hip_runtime.h>
#include <hip/hip_bf16.h>

// Problem dims (fixed by reference)
#define B_   32
#define T_   256
#define IN_  512
#define H_   1024
#define G4_  4096          // 4*H
#define LSTM_BLOCKS 16     // persistent blocks: 128 waves = 2 Mtiles x 64 j-tiles

typedef __bf16 bf16_t;
typedef __attribute__((ext_vector_type(16))) __bf16 v16bf;
typedef __attribute__((ext_vector_type(8)))  __bf16 v8bf;
typedef __attribute__((ext_vector_type(8)))  float  v8f;

// ---------- helpers ----------

__device__ __forceinline__ bf16_t f2bf(float f) {
    // round-to-nearest-even f32 -> bf16
    unsigned u = __builtin_bit_cast(unsigned, f);
    unsigned r = (u + 0x7fffu + ((u >> 16) & 1u)) >> 16;
    unsigned short s = (unsigned short)r;
    return __builtin_bit_cast(bf16_t, s);
}

__device__ __forceinline__ float sigm_(float x) {
    return 1.0f / (1.0f + __expf(-x));
}
__device__ __forceinline__ float tanh_(float x) {
    // tanh(x) = 2*sigmoid(2x) - 1  (stable at both ends)
    return 2.0f * sigm_(2.0f * x) - 1.0f;
}

// A-fragment: 16x32 bf16 tile, row-major source, ld in elements.
// lane L holds row M = L%16; K = (L>=16)*16 + e, 16 contiguous elements
__device__ __forceinline__ v16bf load_a_frag(const bf16_t* tile, int ld, int lane) {
    const bf16_t* p = tile + (size_t)(lane & 15) * ld + ((lane >> 4) << 4);
    union { v16bf v; v8bf h[2]; } u;
    u.h[0] = *(const v8bf*)(p);
    u.h[1] = *(const v8bf*)(p + 8);
    return u.v;
}

// B-fragment: 32x16 (KxN) tile of W^T == 16 rows of row-major W [N,K].
// lane L holds column N = L%16; K = (L>=16)*16 + e, 16 contiguous elements.
__device__ __forceinline__ v16bf load_b_frag(const bf16_t* tile, int ld, int lane) {
    const bf16_t* p = tile + (size_t)(lane & 15) * ld + ((lane >> 4) << 4);
    return *(const v16bf*)p;
}

__device__ __forceinline__ v8f wmma_bf16(v16bf a, v16bf b, v8f c) {
    return __builtin_amdgcn_wmma_f32_16x16x32_bf16(false, a, false, b, (short)0, c, false, false);
}

// monotonic ticket grid barrier (counter must start at a multiple of LSTM_BLOCKS)
__device__ __forceinline__ void grid_sync(unsigned* counter) {
    __threadfence();
    __syncthreads();
    if (threadIdx.x == 0) {
        unsigned ticket = atomicAdd(counter, 1u);
        unsigned target = (ticket / LSTM_BLOCKS + 1u) * LSTM_BLOCKS;
        while (atomicAdd(counter, 0u) < target) {
            __builtin_amdgcn_s_sleep(4);
        }
    }
    __syncthreads();
    __threadfence();
}

// ---------- conversion kernels ----------

__global__ void __launch_bounds__(256)
cvt_kernel(const float* __restrict__ src, bf16_t* __restrict__ dst, int n) {
    for (int i = blockIdx.x * blockDim.x + threadIdx.x; i < n; i += gridDim.x * blockDim.x)
        dst[i] = f2bf(src[i]);
}

__global__ void __launch_bounds__(256)
cvt_scale_kernel(const float* __restrict__ src, const float* __restrict__ alpha,
                 bf16_t* __restrict__ dst, int n) {
    float a = alpha[0];
    for (int i = blockIdx.x * blockDim.x + threadIdx.x; i < n; i += gridDim.x * blockDim.x)
        dst[i] = f2bf(src[i] * a);
}

// ---------- big GEMM: out[M,4096] = A[M,K] x W[4096,K]^T + bias1 + bias2 ----------
// one wave computes a 32x64 macro-tile (2 M-tiles x 4 N-tiles, 8 WMMAs per K-step)

__global__ void __launch_bounds__(256)
gemm_bias_kernel(const bf16_t* __restrict__ A, const bf16_t* __restrict__ W,
                 const float* __restrict__ bias1, const float* __restrict__ bias2,
                 float* __restrict__ out, int K) {
    int lane  = (int)threadIdx.x & 31;
    int gwave = (int)(blockIdx.x * blockDim.x + threadIdx.x) >> 5;
    int wn = gwave & 63;            // 64 column groups of width 64
    int wm = gwave >> 6;            // M/32 row groups
    int m0 = wm * 32;
    int n0 = wn * 64;

    v8f acc[2][4] = {};
    const bf16_t* Arow0 = A + (size_t)m0 * K;
    const bf16_t* Arow1 = A + (size_t)(m0 + 16) * K;

    for (int k0 = 0; k0 < K; k0 += 32) {
        if (k0 + 32 < K) {
            __builtin_prefetch(Arow0 + k0 + 32, 0, 1);
            __builtin_prefetch(Arow1 + k0 + 32, 0, 1);
        }
        v16bf a0 = load_a_frag(Arow0 + k0, K, lane);
        v16bf a1 = load_a_frag(Arow1 + k0, K, lane);
#pragma unroll
        for (int j = 0; j < 4; ++j) {
            const bf16_t* Wt = W + (size_t)(n0 + j * 16) * K + k0;
            v16bf b = load_b_frag(Wt, K, lane);
            acc[0][j] = wmma_bf16(a0, b, acc[0][j]);
            acc[1][j] = wmma_bf16(a1, b, acc[1][j]);
        }
    }

    int nlo   = lane & 15;
    int mhalf = (lane >> 4) << 3;
#pragma unroll
    for (int j = 0; j < 4; ++j) {
        int n = n0 + j * 16 + nlo;
        float bb = bias1[n] + bias2[n];
#pragma unroll
        for (int i = 0; i < 2; ++i) {
            int mb = m0 + i * 16 + mhalf;
#pragma unroll
            for (int r = 0; r < 8; ++r)
                out[(size_t)(mb + r) * G4_ + n] = acc[i][j][r] + bb;
        }
    }
}

// ---------- persistent recurrence kernel (one layer, all 256 timesteps) ----------
// 16 blocks x 8 waves = 128 waves. Each wave owns a 16(batch) x 16(hidden) block of h
// and computes ALL FOUR gate tiles for it (N offsets {j, j+H, j+2H, j+3H}), so the
// full LSTM cell is evaluated on register-resident accumulators: no gate scratch,
// c state lives in registers across all timesteps, ONE grid barrier per step.
// hb is double-buffered (ping-pong) so cell writes never race concurrent GEMM reads.

__global__ void __launch_bounds__(256, 1)
lstm_layer_kernel(const float* __restrict__ G, const bf16_t* __restrict__ Whh,
                  float* __restrict__ hfin, bf16_t* __restrict__ hb2 /* [2][B_*H_] */,
                  float* __restrict__ cfin, bf16_t* __restrict__ ybf,
                  float* __restrict__ yf, unsigned* __restrict__ counter) {
    const int lane  = (int)threadIdx.x & 31;
    const int wave  = (int)threadIdx.x >> 5;
    const int gwave = (int)blockIdx.x * 8 + wave;   // 0..127
    const int jt = gwave & 63;                      // hidden-column tile (64 x 16 cols)
    const int mt = gwave >> 6;                      // batch-row tile (0..1)
    const int nlo = lane & 15;
    const size_t BH = (size_t)B_ * H_;

    // t-invariant per-lane fragment offsets
    const size_t aoff = (size_t)(mt * 16 + nlo) * H_ + ((lane >> 4) << 4);
    const bf16_t* bptr[4];
#pragma unroll
    for (int g = 0; g < 4; ++g)
        bptr[g] = Whh + (size_t)(g * H_ + jt * 16 + nlo) * H_ + ((lane >> 4) << 4);

    const int j  = jt * 16 + nlo;                   // hidden index this lane owns
    const int m0 = mt * 16 + ((lane >> 4) << 3);    // first of 8 batch rows this lane owns

    float creg[8];
#pragma unroll
    for (int r = 0; r < 8; ++r) creg[r] = 0.0f;

    for (int t = 0; t < T_; ++t) {
        const bf16_t* ap = hb2 + (size_t)(t & 1) * BH + aoff;     // read h_{t-1}
        bf16_t*       hw = hb2 + (size_t)((t + 1) & 1) * BH;      // write h_t

        v8f acc[4] = {};                             // i, f, g, o gate h-terms
#pragma unroll 4
        for (int k0 = 0; k0 < H_; k0 += 32) {
            union { v16bf v; v8bf h[2]; } ua;
            ua.h[0] = *(const v8bf*)(ap + k0);
            ua.h[1] = *(const v8bf*)(ap + k0 + 8);
#pragma unroll
            for (int g = 0; g < 4; ++g) {
                v16bf b = *(const v16bf*)(bptr[g] + k0);
                acc[g] = wmma_bf16(ua.v, b, acc[g]);
            }
        }

        // fused LSTM cell on register accumulators (lane = column j, reg r = batch row)
#pragma unroll
        for (int r = 0; r < 8; ++r) {
            int m = m0 + r;
            const float* gp = G + ((size_t)(m * T_ + t)) * G4_;   // x-part + biases
            float vi = gp[j]          + acc[0][r];
            float vf = gp[j + H_]     + acc[1][r];
            float vg = gp[j + 2 * H_] + acc[2][r];
            float vo = gp[j + 3 * H_] + acc[3][r];
            float cn = sigm_(vf) * creg[r] + sigm_(vi) * tanh_(vg);
            float hn = sigm_(vo) * tanh_(cn);
            creg[r] = cn;
            hw[(size_t)m * H_ + j] = f2bf(hn);
            size_t yi = ((size_t)(m * T_ + t)) * H_ + j;
            if (ybf) ybf[yi] = f2bf(hn);    // layer-0 output feeds layer-1 GEMM
            if (yf)  yf[yi]  = hn;          // layer-1 output goes to d_out
            if (t == T_ - 1) hfin[(size_t)m * H_ + j] = hn;
        }

        grid_sync(counter);   // h_t visible to every wave before step t+1's GEMM
    }

#pragma unroll
    for (int r = 0; r < 8; ++r)
        cfin[(size_t)(m0 + r) * H_ + j] = creg[r];
}

// ---------- write h_final / c_final ----------

__global__ void __launch_bounds__(256)
finalize_kernel(const float* __restrict__ h0, const float* __restrict__ h1,
                const float* __restrict__ c0, const float* __restrict__ c1,
                float* __restrict__ out) {
    int i = (int)(blockIdx.x * blockDim.x + threadIdx.x);   // 32768 threads
    float* p = out + (size_t)B_ * T_ * H_;
    p[i]          = h0[i];
    p[32768 + i]  = h1[i];
    p[65536 + i]  = c0[i];
    p[98304 + i]  = c1[i];
}

// ---------- host ----------

extern "C" void kernel_launch(void* const* d_in, const int* in_sizes, int n_in,
                              void* d_out, int out_size, void* d_ws, size_t ws_size,
                              hipStream_t stream) {
    (void)in_sizes; (void)n_in; (void)out_size; (void)ws_size;
    const float* x     = (const float*)d_in[0];
    const float* w_ih0 = (const float*)d_in[1];
    const float* b_ih0 = (const float*)d_in[2];
    const float* a_ih0 = (const float*)d_in[3];
    const float* w_hh0 = (const float*)d_in[4];
    const float* b_hh0 = (const float*)d_in[5];
    const float* a_hh0 = (const float*)d_in[6];
    const float* w_ih1 = (const float*)d_in[7];
    const float* b_ih1 = (const float*)d_in[8];
    const float* a_ih1 = (const float*)d_in[9];
    const float* w_hh1 = (const float*)d_in[10];
    const float* b_hh1 = (const float*)d_in[11];
    const float* a_hh1 = (const float*)d_in[12];
    float* out = (float*)d_out;

    char* ws = (char*)d_ws;
    size_t off = 0;
    auto alloc = [&](size_t bytes) -> char* {
        char* p = ws + off;
        off = (off + bytes + 255) & ~(size_t)255;
        return p;
    };

    float*  Gbuf  = (float*) alloc((size_t)B_ * T_ * G4_ * 4);   // 128 MB (reused by both layers)
    bf16_t* Xb    = (bf16_t*)alloc((size_t)B_ * T_ * IN_ * 2);   // 8 MB
    bf16_t* Y0    = (bf16_t*)alloc((size_t)B_ * T_ * H_  * 2);   // 16 MB
    bf16_t* Wih0b = (bf16_t*)alloc((size_t)G4_ * IN_ * 2);       // 4 MB
    bf16_t* Whh0b = (bf16_t*)alloc((size_t)G4_ * H_  * 2);       // 8 MB
    bf16_t* Wih1b = (bf16_t*)alloc((size_t)G4_ * H_  * 2);       // 8 MB
    bf16_t* Whh1b = (bf16_t*)alloc((size_t)G4_ * H_  * 2);       // 8 MB

    char* stateBegin = ws + off;
    float*    h0s  = (float*) alloc((size_t)B_ * H_ * 4);
    float*    c0s  = (float*) alloc((size_t)B_ * H_ * 4);
    bf16_t*   hb0  = (bf16_t*)alloc((size_t)2 * B_ * H_ * 2);    // ping-pong
    float*    h1s  = (float*) alloc((size_t)B_ * H_ * 4);
    float*    c1s  = (float*) alloc((size_t)B_ * H_ * 4);
    bf16_t*   hb1  = (bf16_t*)alloc((size_t)2 * B_ * H_ * 2);    // ping-pong
    unsigned* ctr  = (unsigned*)alloc(256);
    size_t stateBytes = (size_t)((ws + off) - stateBegin);

    // zero h ping-pong buffers + barrier counter (graph-capturable)
    hipMemsetAsync(stateBegin, 0, stateBytes, stream);

    // bf16 conversions (weights pre-scaled by alpha)
    cvt_kernel      <<<1024, 256, 0, stream>>>(x, Xb, B_ * T_ * IN_);
    cvt_scale_kernel<<< 512, 256, 0, stream>>>(w_ih0, a_ih0, Wih0b, G4_ * IN_);
    cvt_scale_kernel<<<1024, 256, 0, stream>>>(w_hh0, a_hh0, Whh0b, G4_ * H_);
    cvt_scale_kernel<<<1024, 256, 0, stream>>>(w_ih1, a_ih1, Wih1b, G4_ * H_);
    cvt_scale_kernel<<<1024, 256, 0, stream>>>(w_hh1, a_hh1, Whh1b, G4_ * H_);

    // Phase A: G = X @ Wih0^T + b_ih0 + b_hh0   (M=8192, K=512)
    gemm_bias_kernel<<<2048, 256, 0, stream>>>(Xb, Wih0b, b_ih0, b_hh0, Gbuf, IN_);

    // Layer 0 recurrence (persistent, fused gates+cell, writes Y0 bf16)
    lstm_layer_kernel<<<LSTM_BLOCKS, 256, 0, stream>>>(Gbuf, Whh0b, h0s, hb0, c0s,
                                                       Y0, nullptr, ctr);

    // Phase C: G = Y0 @ Wih1^T + b_ih1 + b_hh1  (M=8192, K=1024)
    gemm_bias_kernel<<<2048, 256, 0, stream>>>(Y0, Wih1b, b_ih1, b_hh1, Gbuf, H_);

    // Layer 1 recurrence (persistent, writes out f32) — counter continues from a
    // multiple of LSTM_BLOCKS, so the ticket barrier stays consistent.
    lstm_layer_kernel<<<LSTM_BLOCKS, 256, 0, stream>>>(Gbuf, Whh1b, h1s, hb1, c1s,
                                                       nullptr, out, ctr);

    // h_final / c_final
    finalize_kernel<<<128, 256, 0, stream>>>(h0s, h1s, c0s, c1s, out);
}
